// HMultiHeadAttention_46712064312110
// MI455X (gfx1250) — matrix operand order repaired and checked by reading, not compiled, not run on hardware
//
#include <hip/hip_runtime.h>
#include <hip/hip_bf16.h>
#include <math.h>

// ---------------------------------------------------------------------------
// MI455X (gfx1250, wave32) multi-head attention forward.
// bf16 WMMA (v_wmma_f32_16x16x32_bf16) + fp32 accumulation + fp32 softmax.
// Weights pre-converted to bf16 once (8 MB, L2-resident) so GEMM W-tiles can
// be staged with GLOBAL_LOAD_ASYNC_TO_LDS_B128 (ASYNCcnt path, no VGPR trip).
// d_out = [ out : 8*1024*1024 f32 ][ attn : 8*16*1024*1024 f32 ]
// d_ws  = [Qb 16M][Kb 16M][Vb 16M][Cx 16M][WqB 2M][WkB 2M][WvB 2M][WoB 2M]
// ---------------------------------------------------------------------------

typedef __attribute__((ext_vector_type(16))) __bf16 v16bf;
typedef __attribute__((ext_vector_type(8)))  __bf16 v8bf;
typedef __attribute__((ext_vector_type(8)))  float  v8f;
typedef __attribute__((ext_vector_type(4)))  int    v4i;

#define AS1 __attribute__((address_space(1)))
#define AS3 __attribute__((address_space(3)))

__device__ __forceinline__ __bf16 f2bf(float x) { return (__bf16)x; }

__device__ __forceinline__ v8f v8f_zero() {
  v8f z;
  #pragma unroll
  for (int i = 0; i < 8; ++i) z[i] = 0.0f;
  return z;
}

__device__ __forceinline__ v8f wmma_bf16(v16bf a, v16bf b, v8f c) {
  // D = A(16x32 bf16) * B(32x16 bf16) + C(16x16 f32)
  return __builtin_amdgcn_wmma_f32_16x16x32_bf16(
      /*neg_a=*/false, a, /*neg_b=*/false, b,
      /*c_mod=*/(short)0, c, /*reuse_a=*/false, /*reuse_b=*/false);
}

// A-fragment K index pattern (contiguous, matches B fills): lane half hh picks
// K = 16*hh + 2*i for pair i.
__device__ __forceinline__ int a_kidx(int i, int hh) {
  return 16 * hh + 2 * i;
}

__device__ __forceinline__ void store_out(float*  p, float v) { *p = v; }
__device__ __forceinline__ void store_out(__bf16* p, float v) { *p = f2bf(v); }

// ---------------------------------------------------------------------------
// One-shot fp32 -> bf16 conversion (weights). 8 elems/thread, 16B stores.
// ---------------------------------------------------------------------------
__global__ __launch_bounds__(256)
void cvt_bf16_kernel(const float* __restrict__ src, __bf16* __restrict__ dst, int n) {
  const int i = (blockIdx.x * 256 + threadIdx.x) * 8;
  if (i + 8 <= n) {
    v8bf p;
    #pragma unroll
    for (int u = 0; u < 8; ++u) p[u] = f2bf(src[i + u]);
    *(v8bf*)(dst + i) = p;
  }
}

// ---------------------------------------------------------------------------
// GEMM + bias: C[M,N] = A[M,K](AT) * Wb[K,N](bf16) + bias[N]
// Block tile 64(M) x 128(N), K-step 32. 256 threads = 8 waves in 2x4 grid,
// each wave owns 32x32 = 2x2 WMMA accumulators.
// A tile: VGPR convert + one 16B LDS store per thread.
// W tile: GLOBAL_LOAD_ASYNC_TO_LDS_B128 (2 per thread), s_wait_asynccnt.
// ---------------------------------------------------------------------------
template <typename AT, typename OT>
__global__ __launch_bounds__(256)
void gemm_bias_kernel(const AT* __restrict__ A, const __bf16* __restrict__ Wb,
                      const float* __restrict__ bias, OT* __restrict__ C,
                      int M, int N, int K) {
  // strides: 40*2=80B (16B-aligned, 16 frag rows -> 16 distinct banks),
  //          136*2=272B (=17*16B aligned, 4-bank-group spread)
  __shared__ __align__(16) __bf16 sA[64][40];
  __shared__ __align__(16) __bf16 sB[32][136];

  const int t    = threadIdx.x;
  const int lane = t & 31;
  const int wave = t >> 5;
  const int wm   = wave >> 2;   // 0..1  (M direction)
  const int wn   = wave & 3;    // 0..3  (N direction)
  const int hh   = lane >> 4;   // lane half
  const int ln   = lane & 15;
  const int M0   = blockIdx.x * 64;
  const int N0   = blockIdx.y * 128;

  v8f acc[2][2];
  #pragma unroll
  for (int i = 0; i < 2; ++i)
    #pragma unroll
    for (int j = 0; j < 2; ++j) acc[i][j] = v8f_zero();

  const int ar = t >> 2, ac = (t & 3) * 8;      // A tile 64x32: 8 elems/thread
  const int wrow = t >> 4, wcol = (t & 15) * 8; // W tile 32x128: 2x8 elems/thread

  for (int k0 = 0; k0 < K; k0 += 32) {
    // --- W tile: direct global->LDS async copy (bf16, no conversion) -------
    #pragma unroll
    for (int it = 0; it < 2; ++it) {
      const int row = it * 16 + wrow;
      const __bf16* gsrc = Wb + (size_t)(k0 + row) * N + N0 + wcol;
#if __has_builtin(__builtin_amdgcn_global_load_async_to_lds_b128)
      __builtin_amdgcn_global_load_async_to_lds_b128(
          (AS1 v4i*)gsrc, (AS3 v4i*)&sB[row][wcol], 0, 0);
#else
      *(v8bf*)&sB[row][wcol] = *(const v8bf*)gsrc;
#endif
    }
    // --- A tile: load fp32, convert, one 16B LDS store ---------------------
    {
      v8bf apack;
      #pragma unroll
      for (int u = 0; u < 8; ++u)
        apack[u] = f2bf((float)A[(size_t)(M0 + ar) * K + k0 + ac + u]);
      *(v8bf*)&sA[ar][ac] = apack;
    }
#if __has_builtin(__builtin_amdgcn_global_load_async_to_lds_b128)
    asm volatile("s_wait_asynccnt 0x0" ::: "memory");
#endif
    __syncthreads();

    v16bf afr[2], bfr[2];
    #pragma unroll
    for (int mt = 0; mt < 2; ++mt) {
      const int r = wm * 32 + mt * 16 + ln;
      #pragma unroll
      for (int i = 0; i < 8; ++i) {
        const int kf = a_kidx(i, hh);
        afr[mt][2 * i]     = sA[r][kf];
        afr[mt][2 * i + 1] = sA[r][kf + 1];
      }
    }
    #pragma unroll
    for (int nt = 0; nt < 2; ++nt) {
      const int cn = wn * 32 + nt * 16 + ln;
      #pragma unroll
      for (int j = 0; j < 8; ++j) {
        const int kf = 16 * hh + 2 * j;   // B 32x16: lanes 0-15 K=0..15, 16-31 K=16..31
        bfr[nt][2 * j]     = sB[kf][cn];
        bfr[nt][2 * j + 1] = sB[kf + 1][cn];
      }
    }
    #pragma unroll
    for (int mt = 0; mt < 2; ++mt)
      #pragma unroll
      for (int nt = 0; nt < 2; ++nt)
        acc[mt][nt] = wmma_bf16(afr[mt], bfr[nt], acc[mt][nt]);
    __syncthreads();
  }

  #pragma unroll
  for (int mt = 0; mt < 2; ++mt)
    #pragma unroll
    for (int nt = 0; nt < 2; ++nt) {
      const int col = N0 + wn * 32 + nt * 16 + ln;
      const float bcol = bias[col];
      #pragma unroll
      for (int r = 0; r < 8; ++r) {
        const int row = M0 + wm * 32 + mt * 16 + r + 8 * hh;
        store_out(&C[(size_t)row * N + col], acc[mt][nt][r] + bcol);
      }
    }
}

// ---------------------------------------------------------------------------
// Attention: one workgroup (128 thr = 4 waves) per (b, h, 16-row q tile).
// Phase 1: scores 16x1024 via WMMA -> LDS (fp32, scaled + mask).
// Phase 2: fp32 softmax in LDS; probs -> d_out attn (mandatory 537 MB write).
// Phase 3: ctx = probs * V via WMMA (probs re-read from LDS as bf16 frags).
// ---------------------------------------------------------------------------
#define SSTRIDE 1032   // 1032 % 64 == 8 -> <=2-way bank conflicts in softmax

__global__ __launch_bounds__(128)
void attn_kernel(const __bf16* __restrict__ Qb, const __bf16* __restrict__ Kb,
                 const __bf16* __restrict__ Vb, const float* __restrict__ mask,
                 float* __restrict__ attnOut, __bf16* __restrict__ ctxOut) {
  extern __shared__ float sS[];          // [16][SSTRIDE] fp32 scores/probs (66 KB)
  __shared__ float sRed[16][8];

  const int t    = threadIdx.x;
  const int lane = t & 31;
  const int wave = t >> 5;               // 4 waves
  const int hh   = lane >> 4;
  const int ln   = lane & 15;

  const int wg = blockIdx.x;             // B*H*(S/16) = 8192
  const int qt = wg & 63;
  const int bh = wg >> 6;
  const int h  = bh & 15;
  const int b  = bh >> 4;
  const int q0 = qt * 16;

  // --- Q A-fragments (16 rows x 64 feat = 2 frags), same for all waves ------
  const __bf16* Qbase = Qb + ((size_t)(b * 1024 + q0)) * 1024 + h * 64;
  v16bf aq[2];
  #pragma unroll
  for (int kk = 0; kk < 2; ++kk) {
    const __bf16* qrow = Qbase + (size_t)ln * 1024;
    #pragma unroll
    for (int i = 0; i < 8; ++i) {
      const int kf = kk * 32 + a_kidx(i, hh);
      aq[kk][2 * i]     = qrow[kf];
      aq[kk][2 * i + 1] = qrow[kf + 1];
    }
  }

  // --- Phase 1: scores. Wave w covers key columns [w*256, w*256+256) --------
  const float scale = 0.125f;            // 1/sqrt(64)
  for (int kt = 0; kt < 16; ++kt) {
    const int c0 = wave * 256 + kt * 16;
    v16bf bk0, bk1;
    const __bf16* krow = Kb + ((size_t)(b * 1024 + c0 + ln)) * 1024 + h * 64;
    #pragma unroll
    for (int j = 0; j < 8; ++j) {
      const int kf = 16 * hh + 2 * j;
      bk0[2 * j]     = krow[kf];       bk0[2 * j + 1] = krow[kf + 1];
      bk1[2 * j]     = krow[32 + kf];  bk1[2 * j + 1] = krow[32 + kf + 1];
    }
    v8f c = v8f_zero();
    c = wmma_bf16(aq[0], bk0, c);
    c = wmma_bf16(aq[1], bk1, c);
    #pragma unroll
    for (int r = 0; r < 8; ++r) {
      const int qr  = r + 8 * hh;
      const int col = c0 + ln;
      const float mval = mask[((size_t)(b * 1024 + q0 + qr)) * 1024 + col];
      sS[qr * SSTRIDE + col] = c[r] * scale + mval;
    }
  }
  __syncthreads();

  // --- Phase 2: softmax. 8 threads per row, strided cols --------------------
  const int row = t >> 3;
  const int sl  = t & 7;
  float m = -3.0e38f;
  for (int i = 0; i < 128; ++i) m = fmaxf(m, sS[row * SSTRIDE + sl + 8 * i]);
  sRed[row][sl] = m;
  __syncthreads();
  if (sl == 0) {
    float mm = sRed[row][0];
    #pragma unroll
    for (int i = 1; i < 8; ++i) mm = fmaxf(mm, sRed[row][i]);
    sRed[row][0] = mm;
  }
  __syncthreads();
  const float rowmax = sRed[row][0];
  __syncthreads();                       // sRed reused for sums

  float s = 0.0f;
  for (int i = 0; i < 128; ++i) {
    const int c = sl + 8 * i;
    const float e = __expf(sS[row * SSTRIDE + c] - rowmax);
    sS[row * SSTRIDE + c] = e;
    s += e;
  }
  sRed[row][sl] = s;
  __syncthreads();
  if (sl == 0) {
    float ss = sRed[row][0];
    #pragma unroll
    for (int i = 1; i < 8; ++i) ss += sRed[row][i];
    sRed[row][0] = ss;
  }
  __syncthreads();
  const float inv = 1.0f / sRed[row][0];

  float* attRow = attnOut + ((size_t)((b * 16 + h) * 1024 + q0 + row)) * 1024;
  for (int i = 0; i < 128; ++i) {
    const int c = sl + 8 * i;
    const float p = sS[row * SSTRIDE + c] * inv;
    sS[row * SSTRIDE + c] = p;
    attRow[c] = p;
  }
  __syncthreads();

  // --- Phase 3: ctx = probs(16x1024) * V(1024x64). Wave w owns V cols w*16 --
  const int n0 = wave * 16;
  v8f cacc = v8f_zero();
  for (int kb = 0; kb < 32; ++kb) {
    v16bf ap, bvf;
    #pragma unroll
    for (int i = 0; i < 8; ++i) {
      const int kf = kb * 32 + a_kidx(i, hh);
      ap[2 * i]     = f2bf(sS[ln * SSTRIDE + kf]);
      ap[2 * i + 1] = f2bf(sS[ln * SSTRIDE + kf + 1]);
    }
    #pragma unroll
    for (int j = 0; j < 8; ++j) {
      const int key = kb * 32 + 16 * hh + 2 * j;
      const __bf16* vrow = Vb + ((size_t)(b * 1024 + key)) * 1024 + h * 64 + n0 + ln;
      bvf[2 * j]     = vrow[0];
      bvf[2 * j + 1] = vrow[1024];       // next key row
    }
    cacc = wmma_bf16(ap, bvf, cacc);
  }
  #pragma unroll
  for (int r = 0; r < 8; ++r) {
    const int qr = r + 8 * hh;
    ctxOut[((size_t)(b * 1024 + q0 + qr)) * 1024 + h * 64 + n0 + ln] = f2bf(cacc[r]);
  }
}

// ---------------------------------------------------------------------------
extern "C" void kernel_launch(void* const* d_in, const int* in_sizes, int n_in,
                              void* d_out, int out_size, void* d_ws, size_t ws_size,
                              hipStream_t stream) {
  const float* q    = (const float*)d_in[0];
  const float* k    = (const float*)d_in[1];
  const float* v    = (const float*)d_in[2];
  const float* mask = (const float*)d_in[3];
  const float* Wq   = (const float*)d_in[4];
  const float* bq   = (const float*)d_in[5];
  const float* Wk   = (const float*)d_in[6];
  const float* bk   = (const float*)d_in[7];
  const float* Wv   = (const float*)d_in[8];
  const float* bv   = (const float*)d_in[9];
  const float* Wo   = (const float*)d_in[10];
  const float* bo   = (const float*)d_in[11];

  float* out  = (float*)d_out;                                  // [8,1024,1024]
  float* attn = (float*)d_out + (size_t)8 * 1024 * 1024;        // [8,16,1024,1024]

  char* ws = (char*)d_ws;                     // 72 MB of bf16 intermediates
  const size_t MB = (size_t)1024 * 1024;
  __bf16* Qb  = (__bf16*)(ws);                                  // [8192,1024]
  __bf16* Kb  = (__bf16*)(ws + 16 * MB);
  __bf16* Vb  = (__bf16*)(ws + 32 * MB);
  __bf16* Cx  = (__bf16*)(ws + 48 * MB);
  __bf16* WqB = (__bf16*)(ws + 64 * MB);                        // [1024,1024]
  __bf16* WkB = (__bf16*)(ws + 66 * MB);
  __bf16* WvB = (__bf16*)(ws + 68 * MB);
  __bf16* WoB = (__bf16*)(ws + 70 * MB);

  const int M = 8192, N = 1024, K = 1024;
  const int nW = N * K;                        // 1M elements per weight matrix
  dim3 cblk(256), cgrd(nW / (256 * 8));
  cvt_bf16_kernel<<<cgrd, cblk, 0, stream>>>(Wq, WqB, nW);
  cvt_bf16_kernel<<<cgrd, cblk, 0, stream>>>(Wk, WkB, nW);
  cvt_bf16_kernel<<<cgrd, cblk, 0, stream>>>(Wv, WvB, nW);
  cvt_bf16_kernel<<<cgrd, cblk, 0, stream>>>(Wo, WoB, nW);

  dim3 gblk(256), ggrd(M / 64, N / 128);
  gemm_bias_kernel<float, __bf16><<<ggrd, gblk, 0, stream>>>(q, WqB, bq, Qb, M, N, K);
  gemm_bias_kernel<float, __bf16><<<ggrd, gblk, 0, stream>>>(k, WkB, bk, Kb, M, N, K);
  gemm_bias_kernel<float, __bf16><<<ggrd, gblk, 0, stream>>>(v, WvB, bv, Vb, M, N, K);

  const size_t smem = (size_t)16 * SSTRIDE * sizeof(float);     // 66 KB < 320 KB/WGP
  (void)hipFuncSetAttribute((const void*)attn_kernel,
                            hipFuncAttributeMaxDynamicSharedMemorySize, (int)smem);
  attn_kernel<<<dim3(8 * 16 * 64), dim3(128), smem, stream>>>(Qb, Kb, Vb, mask, attn, Cx);

  gemm_bias_kernel<__bf16, float><<<ggrd, gblk, 0, stream>>>(Cx, WoB, bo, out, M, N, K);
}
